// BasicTransformerBlock_41489384079851
// MI455X (gfx1250) — matrix operand order repaired and checked by reading, not compiled, not run on hardware
//
#include <hip/hip_runtime.h>
#include <hip/hip_bf16.h>

// ============================================================================
// BasicTransformerBlock (temporal sparse-causal attn + cross attn + GEGLU FF)
// for MI455X / gfx1250.
//
// Roofline: ~165 GFLOP vs a few hundred MB of HBM traffic (23.3 TB/s) ->
// strongly compute-bound. All GEMMs and both attention matmuls run on
// v_wmma_f32_16x16x32_bf16 (bf16 inputs preserve fp32 exponent range,
// f32 accumulation). Attention is flash-style (streaming softmax, scores
// never hit memory). wave32, 8 waves / 256-thread block.
// ============================================================================

#define DEV __device__ __forceinline__

typedef __bf16 bf16_t;
typedef __attribute__((ext_vector_type(16))) __bf16 v16bf;
typedef __attribute__((ext_vector_type(8)))  __bf16 v8bf;
typedef __attribute__((ext_vector_type(8)))  float  v8f;

// ---- problem constants ----
#define BFR   8          // batch*frames
#define SEQ   1024
#define DIM   640
#define NH    8
#define DH    80
#define DHP   96         // head dim padded to K-multiple of 32
#define CRS   768
#define LC    77
#define LCP   128        // cross keys padded (2 x 64-key tiles)
#define FI    2560
#define FI2   5120
#define MTOK  (BFR*SEQ)  // 8192
#define MENC  (BFR*LC)   // 616
#define SKV   (2*SEQ)    // 2048
#define ATTN_SCALE 0.1118033988749895f   // 80^-0.5

DEV v8f zero8() {
  v8f z;
#pragma unroll
  for (int i = 0; i < 8; i++) z[i] = 0.f;
  return z;
}

// A fragment (16x32 bf16, M rows x K): lanes 0-15 hold M=lane, K {k0..k0+7,
// k0+16..k0+23}; lanes 16-31 hold K {k0+8..15, k0+24..31}. Two b128 loads.
DEV v16bf load_a_frag(const bf16_t* base, int stride) {
  const int lane = threadIdx.x & 31;
  const int r = lane & 15, hf = lane >> 4;
  const bf16_t* p = base + (size_t)r * stride + hf * 8;
  v8bf lo = *reinterpret_cast<const v8bf*>(p);
  v8bf hi = *reinterpret_cast<const v8bf*>(p + 16);
  v16bf out;
#pragma unroll
  for (int i = 0; i < 8; i++) { out[i] = lo[i]; out[i + 8] = hi[i]; }
  return out;
}

// B fragment (32x16 bf16, K x N cols): lane n holds column n with K
// {k0+8h..+7, k0+16+8h..+7} (same order as A); source layout is [N][K].
DEV v16bf load_b_frag(const bf16_t* base, int stride) {
  const int lane = threadIdx.x & 31;
  const int r = lane & 15, hf = lane >> 4;
  const bf16_t* p = base + (size_t)r * stride + hf * 8;
  v8bf lo = *reinterpret_cast<const v8bf*>(p);
  v8bf hi = *reinterpret_cast<const v8bf*>(p + 16);
  v16bf out;
#pragma unroll
  for (int i = 0; i < 8; i++) { out[i] = lo[i]; out[i + 8] = hi[i]; }
  return out;
}

DEV v8f wmma_bf(v16bf a, v16bf b, v8f c) {
  return __builtin_amdgcn_wmma_f32_16x16x32_bf16(
      /*neg_a=*/false, a, /*neg_b=*/false, b,
      /*c_mod=*/(short)0, c, /*reuse_a=*/false, /*reuse_b=*/false);
}

// ============================================================================
// GEMM: out[M][N] = A[M][K](bf16) * Bt[N][K](bf16)^T (+bias[N]) (+resid f32)
// Block: 128(M) x 64(N), 8 waves, each wave a 32x32 tile (2x2 WMMA frags).
// ============================================================================
template <int OUT_BF16, int RESID>
__global__ __launch_bounds__(256) void gemm_bt_kernel(
    const bf16_t* __restrict__ A, const bf16_t* __restrict__ Bt,
    const float* __restrict__ bias, const float* __restrict__ resid,
    void* __restrict__ outv, int M, int N, int K) {
  const int wave = threadIdx.x >> 5;
  const int lane = threadIdx.x & 31;
  const int r = lane & 15, hf = lane >> 4;
  const int tileM = blockIdx.x * 128 + (wave >> 1) * 32;
  const int tileN = blockIdx.y * 64 + (wave & 1) * 32;

  v8f c00 = zero8(), c01 = zero8(), c10 = zero8(), c11 = zero8();
  const bf16_t* a0p = A + (size_t)tileM * K;
  const bf16_t* a1p = A + (size_t)(tileM + 16) * K;
  const bf16_t* b0p = Bt + (size_t)tileN * K;
  const bf16_t* b1p = Bt + (size_t)(tileN + 16) * K;

  for (int k0 = 0; k0 < K; k0 += 32) {
    if (k0 + 128 < K) {  // emits global_prefetch_b8
      __builtin_prefetch(a0p + k0 + 128, 0, 1);
      __builtin_prefetch(b0p + k0 + 128, 0, 1);
    }
    v16bf a0 = load_a_frag(a0p + k0, K);
    v16bf a1 = load_a_frag(a1p + k0, K);
    v16bf b0 = load_b_frag(b0p + k0, K);
    v16bf b1 = load_b_frag(b1p + k0, K);
    c00 = wmma_bf(a0, b0, c00);
    c01 = wmma_bf(a0, b1, c01);
    c10 = wmma_bf(a1, b0, c10);
    c11 = wmma_bf(a1, b1, c11);
  }

#pragma unroll
  for (int mi = 0; mi < 2; mi++) {
#pragma unroll
    for (int ni = 0; ni < 2; ni++) {
      v8f c = (mi == 0) ? ((ni == 0) ? c00 : c01) : ((ni == 0) ? c10 : c11);
      const int col = tileN + ni * 16 + r;
      const float bv = bias ? bias[col] : 0.f;
#pragma unroll
      for (int e = 0; e < 8; e++) {
        const int row = tileM + mi * 16 + hf * 8 + e;
        if (row < M) {
          float v = c[e] + bv;
          if (RESID) v += resid[(size_t)row * N + col];
          if (OUT_BF16)
            ((bf16_t*)outv)[(size_t)row * N + col] = (bf16_t)v;
          else
            ((float*)outv)[(size_t)row * N + col] = v;
        }
      }
    }
  }
}

// ============================================================================
// Flash attention. Q:[BH][SEQ][DHP] bf16 (dh padded w/ zeros to 96),
// Kc:[BH][Lk][DHP] bf16, Vt:[BH][DH][Lk] bf16 (V transposed).
// Out: merged-head [BFR][SEQ][DIM] bf16. 8 waves x 16 query rows = 128/block.
// ============================================================================
__global__ __launch_bounds__(256) void attn_kernel(
    const bf16_t* __restrict__ Q, const bf16_t* __restrict__ Kc,
    const bf16_t* __restrict__ Vt, bf16_t* __restrict__ Out,
    int Lk, int nvalid, float scale) {
  __shared__ __align__(16) bf16_t pbuf[8][16 * 64];
  const int wave = threadIdx.x >> 5, lane = threadIdx.x & 31;
  const int r = lane & 15, hf = lane >> 4;
  const int bh = blockIdx.y;
  const int h = bh & (NH - 1);
  const int bfi = bh / NH;
  const int row0 = blockIdx.x * 128 + wave * 16;

  const bf16_t* qbase = Q + ((size_t)bh * SEQ + row0) * DHP;
  v16bf a0 = load_a_frag(qbase + 0, DHP);
  v16bf a1 = load_a_frag(qbase + 32, DHP);
  v16bf a2 = load_a_frag(qbase + 64, DHP);

  v8f o[5];
  float m[8], l[8];
#pragma unroll
  for (int d = 0; d < 5; d++) o[d] = zero8();
#pragma unroll
  for (int e = 0; e < 8; e++) { m[e] = -1e30f; l[e] = 0.f; }

  bf16_t* myp = &pbuf[wave][0];

  for (int kt = 0; kt < Lk; kt += 64) {
    // ---- S = scale * Q K^T over a 16x64 key tile (12 WMMAs) ----
    v8f sc[4];
#pragma unroll
    for (int nt = 0; nt < 4; nt++) {
      const bf16_t* kb = Kc + ((size_t)bh * Lk + (kt + nt * 16)) * DHP;
      v8f cc = zero8();
      cc = wmma_bf(a0, load_b_frag(kb + 0, DHP), cc);
      cc = wmma_bf(a1, load_b_frag(kb + 32, DHP), cc);
      cc = wmma_bf(a2, load_b_frag(kb + 64, DHP), cc);
      const int col = kt + nt * 16 + r;
      const float addm = (col < nvalid) ? 0.f : -3.0e38f;
#pragma unroll
      for (int e = 0; e < 8; e++) cc[e] = cc[e] * scale + addm;
      sc[nt] = cc;
    }
    // ---- online softmax (row stats via 16-lane shfl_xor) ----
#pragma unroll
    for (int e = 0; e < 8; e++) {
      float t = fmaxf(fmaxf(sc[0][e], sc[1][e]), fmaxf(sc[2][e], sc[3][e]));
#pragma unroll
      for (int off = 1; off < 16; off <<= 1) t = fmaxf(t, __shfl_xor(t, off, 32));
      const float mn = fmaxf(m[e], t);
      float s = 0.f;
#pragma unroll
      for (int nt = 0; nt < 4; nt++) {
        const float p = __expf(sc[nt][e] - mn);
        s += p;
        myp[(hf * 8 + e) * 64 + nt * 16 + r] = (bf16_t)p;  // C->A relayout via LDS
      }
#pragma unroll
      for (int off = 1; off < 16; off <<= 1) s += __shfl_xor(s, off, 32);
      const float alpha = __expf(m[e] - mn);
      l[e] = l[e] * alpha + s;
      m[e] = mn;
#pragma unroll
      for (int d = 0; d < 5; d++) o[d][e] *= alpha;
    }
    // ---- O += P V (A frags from LDS, B frags from V^T) ----
    v16bf p0 = load_a_frag(myp + 0, 64);
    v16bf p1 = load_a_frag(myp + 32, 64);
#pragma unroll
    for (int d = 0; d < 5; d++) {
      const bf16_t* vb = Vt + ((size_t)bh * DH + d * 16) * Lk + kt;
      o[d] = wmma_bf(p0, load_b_frag(vb + 0, Lk), o[d]);
      o[d] = wmma_bf(p1, load_b_frag(vb + 32, Lk), o[d]);
    }
  }
  // ---- normalize + store merged heads ----
#pragma unroll
  for (int d = 0; d < 5; d++) {
    const int col = h * DH + d * 16 + r;
#pragma unroll
    for (int e = 0; e < 8; e++) {
      const int row = row0 + hf * 8 + e;
      Out[((size_t)bfi * SEQ + row) * DIM + col] = (bf16_t)(o[d][e] / l[e]);
    }
  }
}

// ============================================================================
// LayerNorm over last dim -> bf16
// ============================================================================
__global__ __launch_bounds__(256) void ln_kernel(
    const float* __restrict__ x, const float* __restrict__ w,
    const float* __restrict__ b, bf16_t* __restrict__ y, int Dd) {
  const int row = blockIdx.x;
  const float* xr = x + (size_t)row * Dd;
  float s = 0.f, s2 = 0.f;
  for (int i = threadIdx.x; i < Dd; i += 256) {
    float v = xr[i];
    s += v;
    s2 += v * v;
  }
#pragma unroll
  for (int off = 16; off > 0; off >>= 1) {
    s += __shfl_xor(s, off, 32);
    s2 += __shfl_xor(s2, off, 32);
  }
  __shared__ float rs[8], rs2[8];
  const int wave = threadIdx.x >> 5, lane = threadIdx.x & 31;
  if (lane == 0) { rs[wave] = s; rs2[wave] = s2; }
  __syncthreads();
  if (wave == 0) {
    s = (lane < 8) ? rs[lane] : 0.f;
    s2 = (lane < 8) ? rs2[lane] : 0.f;
#pragma unroll
    for (int off = 4; off > 0; off >>= 1) {
      s += __shfl_xor(s, off, 32);
      s2 += __shfl_xor(s2, off, 32);
    }
    if (lane == 0) { rs[0] = s; rs2[0] = s2; }
  }
  __syncthreads();
  const float mean = rs[0] / Dd;
  const float var = rs2[0] / Dd - mean * mean;
  const float rstd = rsqrtf(var + 1e-5f);
  for (int i = threadIdx.x; i < Dd; i += 256)
    y[(size_t)row * Dd + i] = (bf16_t)((xr[i] - mean) * rstd * w[i] + b[i]);
}

// ============================================================================
// Weight transpose+convert: Wt[n][k] = bf16(W[k][n])  (LDS-tiled)
// ============================================================================
__global__ __launch_bounds__(256) void wtrans_kernel(
    const float* __restrict__ W, bf16_t* __restrict__ Wt, int K, int N) {
  __shared__ float tile[32][33];
  const int kb = blockIdx.x * 32, nb = blockIdx.y * 32;
  const int tx = threadIdx.x & 31, ty = threadIdx.x >> 5;
#pragma unroll
  for (int i = 0; i < 32; i += 8) {
    int k = kb + ty + i, n = nb + tx;
    tile[ty + i][tx] = (k < K && n < N) ? W[(size_t)k * N + n] : 0.f;
  }
  __syncthreads();
#pragma unroll
  for (int i = 0; i < 32; i += 8) {
    int n = nb + ty + i, k = kb + tx;
    if (n < N && k < K) Wt[(size_t)n * K + k] = (bf16_t)tile[tx][ty + i];
  }
}

__global__ __launch_bounds__(256) void f2bf_kernel(const float* __restrict__ x,
                                                   bf16_t* __restrict__ y,
                                                   size_t n) {
  size_t i = (size_t)blockIdx.x * 256 + threadIdx.x;
  if (i < n) y[i] = (bf16_t)x[i];
}

// q pack: [MTOK][DIM] -> [BFR*NH][SEQ][DHP] with zero pad in dh 80..95
__global__ __launch_bounds__(256) void qpack_kernel(
    const bf16_t* __restrict__ qlin, bf16_t* __restrict__ qp) {
  const size_t idx = (size_t)blockIdx.x * 256 + threadIdx.x;
  const size_t total = (size_t)BFR * NH * SEQ * DHP;
  if (idx >= total) return;
  const int d = idx % DHP;
  size_t t = idx / DHP;
  const int srow = t % SEQ;
  t /= SEQ;
  const int h = t % NH;
  const int bfi = t / NH;
  bf16_t v = (bf16_t)0.f;
  if (d < DH) v = qlin[((size_t)bfi * SEQ + srow) * DIM + h * DH + d];
  qp[idx] = v;
}

// sparse-causal gather: kcat[bh][2S][DHP] = [K(frame0) ; K(former)], vt = V^T
__global__ __launch_bounds__(256) void gatherkv_kernel(
    const bf16_t* __restrict__ klin, const bf16_t* __restrict__ vlin,
    bf16_t* __restrict__ kcat, bf16_t* __restrict__ vt,
    const int* __restrict__ vlen) {
  const size_t idx = (size_t)blockIdx.x * 256 + threadIdx.x;
  const size_t total = (size_t)BFR * NH * SKV * DHP;
  if (idx >= total) return;
  const int d = idx % DHP;
  size_t t = idx / DHP;
  const int key = t % SKV;
  t /= SKV;
  const int h = t % NH;
  const int bfi = t / NH;
  const int f = vlen[0];
  const int batch = bfi / f, fi = bfi % f;
  const int part = key >= SEQ;
  const int srow = part ? key - SEQ : key;
  const int sfr = part ? ((fi == 0) ? 0 : fi - 1) : 0;  // former-frame logic
  const int sbf = batch * f + sfr;
  bf16_t kv = (bf16_t)0.f;
  if (d < DH) kv = klin[((size_t)sbf * SEQ + srow) * DIM + h * DH + d];
  kcat[idx] = kv;
  if (d < DH) {
    bf16_t vv = vlin[((size_t)sbf * SEQ + srow) * DIM + h * DH + d];
    vt[(((size_t)bfi * NH + h) * DH + d) * SKV + key] = vv;
  }
}

// cross-attn K/V split-heads + pad: kh[bh][LCP][DHP], vt[bh][DH][LCP]
__global__ __launch_bounds__(256) void gatherkv2_kernel(
    const bf16_t* __restrict__ klin, const bf16_t* __restrict__ vlin,
    bf16_t* __restrict__ kh, bf16_t* __restrict__ vt) {
  const size_t idx = (size_t)blockIdx.x * 256 + threadIdx.x;
  const size_t total = (size_t)BFR * NH * LCP * DHP;
  if (idx >= total) return;
  const int d = idx % DHP;
  size_t t = idx / DHP;
  const int key = t % LCP;
  t /= LCP;
  const int h = t % NH;
  const int bfi = t / NH;
  const bool kval = (key < LC) && (d < DH);
  kh[idx] = kval ? klin[((size_t)bfi * LC + key) * DIM + h * DH + d] : (bf16_t)0.f;
  if (d < DH)
    vt[(((size_t)bfi * NH + h) * DH + d) * LCP + key] =
        (key < LC) ? vlin[((size_t)bfi * LC + key) * DIM + h * DH + d]
                   : (bf16_t)0.f;
}

// GEGLU: g = h * gelu_tanh(gate), proj = [h | gate]
__global__ __launch_bounds__(256) void geglu_kernel(
    const bf16_t* __restrict__ proj, bf16_t* __restrict__ g) {
  const size_t idx = (size_t)blockIdx.x * 256 + threadIdx.x;
  const size_t total = (size_t)MTOK * FI;
  if (idx >= total) return;
  const int j = idx % FI;
  const size_t row = idx / FI;
  const float hh = (float)proj[row * FI2 + j];
  const float gg = (float)proj[row * FI2 + FI + j];
  const float u = gg + 0.044715f * gg * gg * gg;
  const float ge = 0.5f * gg * (1.f + tanhf(0.7978845608028654f * u));
  g[idx] = (bf16_t)(hh * ge);
}

// ============================================================================
// Host launch
// ============================================================================
extern "C" void kernel_launch(void* const* d_in, const int* in_sizes, int n_in,
                              void* d_out, int out_size, void* d_ws,
                              size_t ws_size, hipStream_t stream) {
  (void)in_sizes; (void)n_in; (void)out_size; (void)ws_size;
  const float* hidden = (const float*)d_in[0];
  const float* enc    = (const float*)d_in[1];
  const int*   vlen   = (const int*)d_in[2];
  const float* ln1w = (const float*)d_in[3];
  const float* ln1b = (const float*)d_in[4];
  const float* q1 = (const float*)d_in[5];
  const float* k1 = (const float*)d_in[6];
  const float* v1 = (const float*)d_in[7];
  const float* o1w = (const float*)d_in[8];
  const float* o1b = (const float*)d_in[9];
  const float* ln2w = (const float*)d_in[10];
  const float* ln2b = (const float*)d_in[11];
  const float* q2 = (const float*)d_in[12];
  const float* k2 = (const float*)d_in[13];
  const float* v2 = (const float*)d_in[14];
  const float* o2w = (const float*)d_in[15];
  const float* o2b = (const float*)d_in[16];
  const float* ln3w = (const float*)d_in[17];
  const float* ln3b = (const float*)d_in[18];
  const float* ffw1 = (const float*)d_in[19];
  const float* ffb1 = (const float*)d_in[20];
  const float* ffw2 = (const float*)d_in[21];
  const float* ffb2 = (const float*)d_in[22];

  char* ws = (char*)d_ws;
  size_t off = 0;
  auto alloc = [&](size_t bytes) -> char* {
    char* p = ws + off;
    off += (bytes + 255) & ~(size_t)255;
    return p;
  };
  bf16_t* xln   = (bf16_t*)alloc((size_t)MTOK * DIM * 2);
  bf16_t* wq1t  = (bf16_t*)alloc((size_t)DIM * DIM * 2);
  bf16_t* wk1t  = (bf16_t*)alloc((size_t)DIM * DIM * 2);
  bf16_t* wv1t  = (bf16_t*)alloc((size_t)DIM * DIM * 2);
  bf16_t* wo1t  = (bf16_t*)alloc((size_t)DIM * DIM * 2);
  bf16_t* wq2t  = (bf16_t*)alloc((size_t)DIM * DIM * 2);
  bf16_t* wk2t  = (bf16_t*)alloc((size_t)DIM * CRS * 2);
  bf16_t* wv2t  = (bf16_t*)alloc((size_t)DIM * CRS * 2);
  bf16_t* wo2t  = (bf16_t*)alloc((size_t)DIM * DIM * 2);
  bf16_t* wf1t  = (bf16_t*)alloc((size_t)FI2 * DIM * 2);
  bf16_t* wf2t  = (bf16_t*)alloc((size_t)DIM * FI * 2);
  bf16_t* qlin  = (bf16_t*)alloc((size_t)MTOK * DIM * 2);
  bf16_t* klin  = (bf16_t*)alloc((size_t)MTOK * DIM * 2);
  bf16_t* vlin  = (bf16_t*)alloc((size_t)MTOK * DIM * 2);
  bf16_t* qp    = (bf16_t*)alloc((size_t)BFR * NH * SEQ * DHP * 2);
  bf16_t* kcat  = (bf16_t*)alloc((size_t)BFR * NH * SKV * DHP * 2);
  bf16_t* vt    = (bf16_t*)alloc((size_t)BFR * NH * DH * SKV * 2);
  bf16_t* atto  = (bf16_t*)alloc((size_t)MTOK * DIM * 2);
  float*  hid1  = (float*)alloc((size_t)MTOK * DIM * 4);
  float*  hid2  = (float*)alloc((size_t)MTOK * DIM * 4);
  bf16_t* encb  = (bf16_t*)alloc((size_t)640 * CRS * 2);  // 616 rows + slack
  bf16_t* k2l   = (bf16_t*)alloc((size_t)640 * DIM * 2);
  bf16_t* v2l   = (bf16_t*)alloc((size_t)640 * DIM * 2);
  bf16_t* k2h   = (bf16_t*)alloc((size_t)BFR * NH * LCP * DHP * 2);
  bf16_t* vt2   = (bf16_t*)alloc((size_t)BFR * NH * DH * LCP * 2);
  bf16_t* proj  = (bf16_t*)alloc((size_t)MTOK * FI2 * 2);
  bf16_t* gbuf  = (bf16_t*)alloc((size_t)MTOK * FI * 2);

  const dim3 blk(256);
  auto g1 = [](size_t n) { return dim3((unsigned)((n + 255) / 256)); };

  // ---- weight transposes (fp32 -> bf16, [K][N] -> [N][K]) ----
  wtrans_kernel<<<dim3(DIM / 32, DIM / 32), blk, 0, stream>>>(q1, wq1t, DIM, DIM);
  wtrans_kernel<<<dim3(DIM / 32, DIM / 32), blk, 0, stream>>>(k1, wk1t, DIM, DIM);
  wtrans_kernel<<<dim3(DIM / 32, DIM / 32), blk, 0, stream>>>(v1, wv1t, DIM, DIM);
  wtrans_kernel<<<dim3(DIM / 32, DIM / 32), blk, 0, stream>>>(o1w, wo1t, DIM, DIM);
  wtrans_kernel<<<dim3(DIM / 32, DIM / 32), blk, 0, stream>>>(q2, wq2t, DIM, DIM);
  wtrans_kernel<<<dim3(CRS / 32, DIM / 32), blk, 0, stream>>>(k2, wk2t, CRS, DIM);
  wtrans_kernel<<<dim3(CRS / 32, DIM / 32), blk, 0, stream>>>(v2, wv2t, CRS, DIM);
  wtrans_kernel<<<dim3(DIM / 32, DIM / 32), blk, 0, stream>>>(o2w, wo2t, DIM, DIM);
  wtrans_kernel<<<dim3(DIM / 32, FI2 / 32), blk, 0, stream>>>(ffw1, wf1t, DIM, FI2);
  wtrans_kernel<<<dim3(FI / 32, DIM / 32), blk, 0, stream>>>(ffw2, wf2t, FI, DIM);

  // ---- self attention ----
  ln_kernel<<<dim3(MTOK), blk, 0, stream>>>(hidden, ln1w, ln1b, xln, DIM);
  gemm_bt_kernel<1, 0><<<dim3(MTOK / 128, DIM / 64), blk, 0, stream>>>(
      xln, wq1t, nullptr, nullptr, qlin, MTOK, DIM, DIM);
  gemm_bt_kernel<1, 0><<<dim3(MTOK / 128, DIM / 64), blk, 0, stream>>>(
      xln, wk1t, nullptr, nullptr, klin, MTOK, DIM, DIM);
  gemm_bt_kernel<1, 0><<<dim3(MTOK / 128, DIM / 64), blk, 0, stream>>>(
      xln, wv1t, nullptr, nullptr, vlin, MTOK, DIM, DIM);
  qpack_kernel<<<g1((size_t)BFR * NH * SEQ * DHP), blk, 0, stream>>>(qlin, qp);
  gatherkv_kernel<<<g1((size_t)BFR * NH * SKV * DHP), blk, 0, stream>>>(
      klin, vlin, kcat, vt, vlen);
  attn_kernel<<<dim3(SEQ / 128, BFR * NH), blk, 0, stream>>>(
      qp, kcat, vt, atto, SKV, SKV, ATTN_SCALE);
  gemm_bt_kernel<0, 1><<<dim3(MTOK / 128, DIM / 64), blk, 0, stream>>>(
      atto, wo1t, o1b, hidden, hid1, MTOK, DIM, DIM);

  // ---- cross attention ----
  ln_kernel<<<dim3(MTOK), blk, 0, stream>>>(hid1, ln2w, ln2b, xln, DIM);
  gemm_bt_kernel<1, 0><<<dim3(MTOK / 128, DIM / 64), blk, 0, stream>>>(
      xln, wq2t, nullptr, nullptr, qlin, MTOK, DIM, DIM);
  qpack_kernel<<<g1((size_t)BFR * NH * SEQ * DHP), blk, 0, stream>>>(qlin, qp);
  f2bf_kernel<<<g1((size_t)MENC * CRS), blk, 0, stream>>>(enc, encb,
                                                          (size_t)MENC * CRS);
  gemm_bt_kernel<1, 0><<<dim3((MENC + 127) / 128, DIM / 64), blk, 0, stream>>>(
      encb, wk2t, nullptr, nullptr, k2l, MENC, DIM, CRS);
  gemm_bt_kernel<1, 0><<<dim3((MENC + 127) / 128, DIM / 64), blk, 0, stream>>>(
      encb, wv2t, nullptr, nullptr, v2l, MENC, DIM, CRS);
  gatherkv2_kernel<<<g1((size_t)BFR * NH * LCP * DHP), blk, 0, stream>>>(
      k2l, v2l, k2h, vt2);
  attn_kernel<<<dim3(SEQ / 128, BFR * NH), blk, 0, stream>>>(
      qp, k2h, vt2, atto, LCP, LC, ATTN_SCALE);
  gemm_bt_kernel<0, 1><<<dim3(MTOK / 128, DIM / 64), blk, 0, stream>>>(
      atto, wo2t, o2b, hid1, hid2, MTOK, DIM, DIM);

  // ---- GEGLU feed-forward ----
  ln_kernel<<<dim3(MTOK), blk, 0, stream>>>(hid2, ln3w, ln3b, xln, DIM);
  gemm_bt_kernel<1, 0><<<dim3(MTOK / 128, FI2 / 64), blk, 0, stream>>>(
      xln, wf1t, ffb1, nullptr, proj, MTOK, FI2, DIM);
  geglu_kernel<<<g1((size_t)MTOK * FI), blk, 0, stream>>>(proj, gbuf);
  gemm_bt_kernel<0, 1><<<dim3(MTOK / 128, DIM / 64), blk, 0, stream>>>(
      gbuf, wf2t, ffb2, hid2, (float*)d_out, MTOK, DIM, FI);
}
